// SpectrogramMambaBranch_58832462020950
// MI455X (gfx1250) — matrix operand order repaired and checked by reading, not compiled, not run on hardware
//
#include <hip/hip_runtime.h>
#include <math.h>

typedef __attribute__((ext_vector_type(16))) _Float16 v16h;
typedef __attribute__((ext_vector_type(8)))  float    v8f;

#define D_MODEL 64
#define D_INNER 128
#define D_STATE 16
#define DT_RANK 4
#define NPROJ   36          // DT_RANK + 2*D_STATE
#define BB      8
#define CIN     4
#define FF      128
#define TT      256
#define FTSZ    (FF*TT)
#define NTOK    (BB*FF*TT)  // 262144 tokens
#define EPSF    1e-5f

// ---------------------------------------------------------------------------
// small utility kernels
// ---------------------------------------------------------------------------
__global__ void k_zero(float* __restrict__ p, int n) {
    int i = blockIdx.x * blockDim.x + threadIdx.x;
    if (i < n) p[i] = 0.f;
}

__global__ void k_negexp(float* __restrict__ dst, const float* __restrict__ src, int n) {
    int i = blockIdx.x * blockDim.x + threadIdx.x;
    if (i < n) dst[i] = -__expf(src[i]);
}

// prepack conv2 weights (O=64, I=32, 3, 3) -> tap-major wt[q][n][c]
__global__ void k_packw2(const float* __restrict__ w2, float* __restrict__ wt) {
    int i = blockIdx.x * blockDim.x + threadIdx.x;   // < 9*64*32
    if (i >= 9 * 64 * 32) return;
    int c = i & 31;
    int n = (i >> 5) & 63;
    int q = i >> 11;
    wt[i] = w2[(n * 32 + c) * 9 + q];
}

// ---------------------------------------------------------------------------
// conv1: 4->32, 3x3 SAME, channel-last output (pixel-major, 32 ch contiguous).
// lane = output channel -> coalesced stores, contiguous 36-float weight row.
// ---------------------------------------------------------------------------
__global__ void k_conv1(const float* __restrict__ x, const float* __restrict__ w,
                        float* __restrict__ out) {
    int i   = blockIdx.x * 256 + threadIdx.x;        // < NTOK*32
    int co  = i & 31;
    int pix = i >> 5;
    int t   = pix & 255;
    int f   = (pix >> 8) & 127;
    int b   = pix >> 15;
    // 36 contiguous weights for this output channel: (ci,ky,kx)
    float wreg[36];
    const float4* wv = (const float4*)(w + co * 36);
    #pragma unroll
    for (int j = 0; j < 9; ++j) {
        float4 v = wv[j];
        wreg[4 * j] = v.x; wreg[4 * j + 1] = v.y; wreg[4 * j + 2] = v.z; wreg[4 * j + 3] = v.w;
    }
    float s = 0.f;
    #pragma unroll
    for (int ci = 0; ci < CIN; ++ci) {
        const float* xp = x + ((size_t)(b * CIN + ci) * FF) * TT;
        #pragma unroll
        for (int ky = 0; ky < 3; ++ky) {
            int fy = f + ky - 1;
            if (fy < 0 || fy >= FF) continue;
            #pragma unroll
            for (int kx = 0; kx < 3; ++kx) {
                int tx = t + kx - 1;
                if (tx < 0 || tx >= TT) continue;
                s += xp[fy * TT + tx] * wreg[ci * 9 + ky * 3 + kx];
            }
        }
    }
    out[(size_t)pix * 32 + co] = s;
}

// ---------------------------------------------------------------------------
// per-channel sum/sumsq reduction, token-major buffer (G rows of C channels)
// ---------------------------------------------------------------------------
__global__ void k_stats_tok(const float* __restrict__ buf, int C, int G,
                            float* __restrict__ sum, float* __restrict__ sumsq) {
    int c = blockIdx.x;
    float s = 0.f, ss = 0.f;
    for (int i = threadIdx.x; i < G; i += blockDim.x) {
        float v = buf[(size_t)i * C + c];
        s += v; ss += v * v;
    }
    __shared__ float rs[256], rq[256];
    rs[threadIdx.x] = s; rq[threadIdx.x] = ss;
    __syncthreads();
    for (int o = blockDim.x >> 1; o > 0; o >>= 1) {
        if ((int)threadIdx.x < o) { rs[threadIdx.x] += rs[threadIdx.x + o];
                                    rq[threadIdx.x] += rq[threadIdx.x + o]; }
        __syncthreads();
    }
    if (threadIdx.x == 0) { sum[c] = rs[0]; sumsq[c] = rq[0]; }
}

__global__ void k_bnfin(const float* __restrict__ sum, const float* __restrict__ sumsq,
                        float invN, const float* __restrict__ g, const float* __restrict__ b,
                        float* __restrict__ scale, float* __restrict__ shift, int C) {
    int c = threadIdx.x;
    if (c >= C) return;
    float m   = sum[c] * invN;
    float var = sumsq[c] * invN - m * m;
    float inv = rsqrtf(var + EPSF);
    float sc  = g[c] * inv;
    scale[c] = sc;
    shift[c] = b[c] - m * sc;
}

// ---------------------------------------------------------------------------
// conv2 as implicit GEMM via V_WMMA_F32_16X16X32_F16 (branch-free, b128 feeds).
// M = NTOK (token = b,f,t), N = 64 output channels, K = 9 taps * 32 channels.
// Input channel-last -> per-lane A operand per tap = two contiguous 8-float
// runs; prepacked tap-major weights -> B operand = two contiguous 8-float
// runs. bn1+ReLU folded into the A gather; zero-padding by clamp + 0/1 mask.
// ---------------------------------------------------------------------------
__global__ void __launch_bounds__(256)
k_conv2_wmma(const float* __restrict__ c1, const float* __restrict__ wt,
             const float* __restrict__ sc1, const float* __restrict__ sh1,
             float* __restrict__ out) {
    int wave = threadIdx.x >> 5;
    int lane = threadIdx.x & 31;
    int half = lane >> 4;
    int r    = lane & 15;
    int m0   = (blockIdx.x * 8 + wave) * 16;
    int n0   = blockIdx.y * 16;
    int m    = m0 + r;
    int b    = m >> 15;
    int f    = (m >> 8) & 127;
    int t    = m & 255;

    // per-lane bn1 scale/shift in A-packing order (run1 = half*8+0..7, run2 = +16)
    float scv[16], shv[16];
    #pragma unroll
    for (int j = 0; j < 8; ++j) {
        int c0 = (half << 3) + j;
        int c1c = 16 + (half << 3) + j;
        scv[j]     = sc1[c0];  shv[j]     = sh1[c0];
        scv[8 + j] = sc1[c1c]; shv[8 + j] = sh1[c1c];
    }
    const float* bwt   = wt + (size_t)(n0 + r) * 32 + (half << 3);  // + q*64*32
    const float* cbase = c1 + (size_t)b * FTSZ * 32;

    v8f acc = {};
    #pragma unroll
    for (int q = 0; q < 9; ++q) {            // k-chunk = one 3x3 tap, 32 channels
        int ky = q / 3, kx = q - 3 * (q / 3);
        int fy = f + ky - 1, tx = t + kx - 1;
        float msk = ((fy >= 0) && (fy < FF) && (tx >= 0) && (tx < TT)) ? 1.f : 0.f;
        int fyc = min(max(fy, 0), FF - 1);
        int txc = min(max(tx, 0), TT - 1);
        const float4* s4 = (const float4*)(cbase + ((size_t)fyc * TT + txc) * 32 + (half << 3));
        float4 r0 = s4[0], r1 = s4[1];       // channels half*8 + 0..7
        float4 r2 = s4[4], r3 = s4[5];       // channels 16 + half*8 + 0..7
        const float4* w4 = (const float4*)(bwt + (size_t)q * (64 * 32));
        float4 w0 = w4[0], w1 = w4[1], w2 = w4[4], w3 = w4[5];
        float av[16] = { r0.x, r0.y, r0.z, r0.w, r1.x, r1.y, r1.z, r1.w,
                         r2.x, r2.y, r2.z, r2.w, r3.x, r3.y, r3.z, r3.w };
        float bv[16] = { w0.x, w0.y, w0.z, w0.w, w1.x, w1.y, w1.z, w1.w,
                         w2.x, w2.y, w2.z, w2.w, w3.x, w3.y, w3.z, w3.w };
        v16h a, bm;
        #pragma unroll
        for (int j = 0; j < 16; ++j) {
            a[j]  = (_Float16)(msk * fmaxf(av[j] * scv[j] + shv[j], 0.f));
            bm[j] = (_Float16)bv[j];
        }
        acc = __builtin_amdgcn_wmma_f32_16x16x32_f16(false, a, false, bm,
                                                     (short)0, acc, false, false);
    }
    #pragma unroll
    for (int v = 0; v < 8; ++v)
        out[(size_t)(m0 + v + 8 * half) * 64 + (n0 + r)] = acc[v];
}

// ---------------------------------------------------------------------------
// generic WMMA GEMM: C[M,N] = X[M,K] * W[N,K]^T, f32 I/O, f16 compute.
// Per k-chunk, each lane's A operand is two contiguous 8-float runs and the
// B operand likewise two contiguous 8-float runs -> fetched with global_load_b128.
// K multiple of 32, N multiple of 16, M multiple of 128. One wave per 16x16 tile.
// ---------------------------------------------------------------------------
__global__ void __launch_bounds__(256)
k_gemm_wmma(const float* __restrict__ X, const float* __restrict__ W,
            float* __restrict__ C, int M, int N, int K) {
    int wave = threadIdx.x >> 5;
    int lane = threadIdx.x & 31;
    int half = lane >> 4;
    int r    = lane & 15;
    int m0   = (blockIdx.x * 8 + wave) * 16;
    int n0   = blockIdx.y * 16;
    const float* xrow = X + (size_t)(m0 + r) * K;
    const float* wrow = W + (size_t)(n0 + r) * K;
    v8f acc = {};
    for (int kk = 0; kk < K; kk += 32) {
        // A: k = kk + half*8 + {0..7}  and  kk + 16 + half*8 + {0..7}
        const float4* ax = (const float4*)(xrow + kk + half * 8);
        float4 a0 = ax[0], a1 = ax[1];   // first 8-float run
        float4 a2 = ax[4], a3 = ax[5];   // +16 floats
        // B: k = kk + half*8 + {0..7}  and  kk + 16 + half*8 + {0..7}
        const float4* bx = (const float4*)(wrow + kk + half * 8);
        float4 b0 = bx[0], b1 = bx[1], b2 = bx[4], b3 = bx[5];
        v16h a, bm;
        a[0]  = (_Float16)a0.x; a[1]  = (_Float16)a0.y; a[2]  = (_Float16)a0.z; a[3]  = (_Float16)a0.w;
        a[4]  = (_Float16)a1.x; a[5]  = (_Float16)a1.y; a[6]  = (_Float16)a1.z; a[7]  = (_Float16)a1.w;
        a[8]  = (_Float16)a2.x; a[9]  = (_Float16)a2.y; a[10] = (_Float16)a2.z; a[11] = (_Float16)a2.w;
        a[12] = (_Float16)a3.x; a[13] = (_Float16)a3.y; a[14] = (_Float16)a3.z; a[15] = (_Float16)a3.w;
        bm[0]  = (_Float16)b0.x; bm[1]  = (_Float16)b0.y; bm[2]  = (_Float16)b0.z; bm[3]  = (_Float16)b0.w;
        bm[4]  = (_Float16)b1.x; bm[5]  = (_Float16)b1.y; bm[6]  = (_Float16)b1.z; bm[7]  = (_Float16)b1.w;
        bm[8]  = (_Float16)b2.x; bm[9]  = (_Float16)b2.y; bm[10] = (_Float16)b2.z; bm[11] = (_Float16)b2.w;
        bm[12] = (_Float16)b3.x; bm[13] = (_Float16)b3.y; bm[14] = (_Float16)b3.z; bm[15] = (_Float16)b3.w;
        acc = __builtin_amdgcn_wmma_f32_16x16x32_f16(false, a, false, bm,
                                                     (short)0, acc, false, false);
    }
    #pragma unroll
    for (int v = 0; v < 8; ++v)
        C[(size_t)(m0 + v + 8 * half) * N + (n0 + r)] = acc[v];
}

// ---------------------------------------------------------------------------
// fused bn2(scale/shift)+ReLU+LayerNorm, wave per token (input token-major)
// ---------------------------------------------------------------------------
__global__ void k_bn2ln(const float* __restrict__ c2, const float* __restrict__ sc,
                        const float* __restrict__ sh, const float* __restrict__ g,
                        const float* __restrict__ bt, float* __restrict__ xo) {
    int wave = threadIdx.x >> 5, lane = threadIdx.x & 31;
    size_t tok = (size_t)blockIdx.x * 8 + wave;
    const float* row = c2 + tok * 64;
    int c0 = lane, c1 = lane + 32;
    float v0 = fmaxf(row[c0] * sc[c0] + sh[c0], 0.f);
    float v1 = fmaxf(row[c1] * sc[c1] + sh[c1], 0.f);
    float s = v0 + v1, ss = v0 * v0 + v1 * v1;
    for (int o = 16; o >= 1; o >>= 1) { s += __shfl_xor(s, o); ss += __shfl_xor(ss, o); }
    float m = s * (1.f / 64.f), var = ss * (1.f / 64.f) - m * m;
    float inv = rsqrtf(var + EPSF);
    xo[tok * 64 + c0] = (v0 - m) * inv * g[c0] + bt[c0];
    xo[tok * 64 + c1] = (v1 - m) * inv * g[c1] + bt[c1];
}

// plain LayerNorm, wave per token
__global__ void k_ln(const float* __restrict__ h, const float* __restrict__ g,
                     const float* __restrict__ bt, float* __restrict__ xo) {
    int wave = threadIdx.x >> 5, lane = threadIdx.x & 31;
    size_t tok = (size_t)blockIdx.x * 8 + wave;
    const float* row = h + tok * 64;
    int c0 = lane, c1 = lane + 32;
    float v0 = row[c0], v1 = row[c1];
    float s = v0 + v1, ss = v0 * v0 + v1 * v1;
    for (int o = 16; o >= 1; o >>= 1) { s += __shfl_xor(s, o); ss += __shfl_xor(ss, o); }
    float m = s * (1.f / 64.f), var = ss * (1.f / 64.f) - m * m;
    float inv = rsqrtf(var + EPSF);
    xo[tok * 64 + c0] = (v0 - m) * inv * g[c0] + bt[c0];
    xo[tok * 64 + c1] = (v1 - m) * inv * g[c1] + bt[c1];
}

// ---------------------------------------------------------------------------
// causal depthwise conv1d (k=4) + bias + SiLU, in-place, sliding window in regs
// ---------------------------------------------------------------------------
__global__ void k_dwconv(float* __restrict__ xb, const float* __restrict__ cw,
                         const float* __restrict__ cb, int nseq, int L) {
    int i = blockIdx.x * blockDim.x + threadIdx.x;
    if (i >= nseq * D_INNER) return;
    int d = i & (D_INNER - 1);
    int seq = i >> 7;
    float w0 = cw[d * 4 + 0], w1 = cw[d * 4 + 1], w2 = cw[d * 4 + 2], w3 = cw[d * 4 + 3];
    float bb = cb[d];
    float x3 = 0.f, x2 = 0.f, x1 = 0.f;
    size_t base = (size_t)seq * L * D_INNER + d;
    for (int l = 0; l < L; ++l) {
        float x0 = xb[base + (size_t)l * D_INNER];
        float acc = x3 * w0 + x2 * w1 + x1 * w2 + x0 * w3 + bb;
        float sig = 1.f / (1.f + __expf(-acc));
        xb[base + (size_t)l * D_INNER] = acc * sig;
        x3 = x2; x2 = x1; x1 = x0;
    }
}

// ---------------------------------------------------------------------------
// x_proj: proj[g, 0..35] = xc[g,:] @ x_proj_w^T  (wave per token, b128 loads)
// ---------------------------------------------------------------------------
__global__ void k_xproj(const float* __restrict__ xc, const float* __restrict__ xw,
                        float* __restrict__ proj) {
    int wave = threadIdx.x >> 5, lane = threadIdx.x & 31;
    size_t g = (size_t)blockIdx.x * 8 + wave;
    const float4* xrow = (const float4*)(xc + g * D_INNER);
    for (int j = lane; j < NPROJ; j += 32) {
        const float4* wr = (const float4*)(xw + j * D_INNER);
        float s = 0.f;
        #pragma unroll 8
        for (int k = 0; k < D_INNER / 4; ++k) {
            float4 xv = xrow[k], wv = wr[k];
            s += xv.x * wv.x + xv.y * wv.y + xv.z * wv.z + xv.w * wv.w;
        }
        proj[g * NPROJ + j] = s;
    }
}

// ---------------------------------------------------------------------------
// SSM selective scan: block per sequence, thread per inner channel d.
// h[16] + A-row in registers; per-step proj (dt-lin, B, C) staged in LDS.
// Fuses softplus(dt), D-skip and silu(z) gate; writes in-place over xc.
// ---------------------------------------------------------------------------
__global__ void __launch_bounds__(128)
k_scan(float* __restrict__ xc, const float* __restrict__ proj,
       const float* __restrict__ z, const float* __restrict__ A,
       const float* __restrict__ Dp, const float* __restrict__ dtw,
       const float* __restrict__ dtb, int L) {
    __shared__ float sp[NPROJ];
    int d = threadIdx.x;
    int seq = blockIdx.x;
    float h[D_STATE], a[D_STATE];
    #pragma unroll
    for (int s = 0; s < D_STATE; ++s) { h[s] = 0.f; a[s] = A[d * D_STATE + s]; }
    float w0 = dtw[d * 4 + 0], w1 = dtw[d * 4 + 1], w2 = dtw[d * 4 + 2], w3 = dtw[d * 4 + 3];
    float bd = dtb[d], Dd = Dp[d];
    size_t base = (size_t)seq * L;
    for (int l = 0; l < L; ++l) {
        size_t g = base + l;
        __syncthreads();
        if (threadIdx.x < NPROJ) sp[threadIdx.x] = proj[g * NPROJ + threadIdx.x];
        __syncthreads();
        float u   = xc[g * D_INNER + d];
        float pre = sp[0] * w0 + sp[1] * w1 + sp[2] * w2 + sp[3] * w3 + bd;
        float dt  = (pre > 20.f) ? pre : log1pf(__expf(pre));
        float du  = dt * u;
        float y   = 0.f;
        #pragma unroll
        for (int s = 0; s < D_STATE; ++s) {
            float dA = __expf(dt * a[s]);
            h[s] = h[s] * dA + du * sp[4 + s];
            y += h[s] * sp[20 + s];
        }
        y += u * Dd;
        float zz  = z[g * D_INNER + d];
        float sig = 1.f / (1.f + __expf(-zz));
        xc[g * D_INNER + d] = y * (zz * sig);
    }
}

// ---------------------------------------------------------------------------
// mean over (F,T) per (batch, channel)
// ---------------------------------------------------------------------------
__global__ void k_mean(const float* __restrict__ hb, float* __restrict__ mean) {
    int b = blockIdx.x, c = blockIdx.y;
    int per = FF * TT;
    size_t base = (size_t)b * per;
    float s = 0.f;
    for (int i = threadIdx.x; i < per; i += blockDim.x) s += hb[(base + i) * 64 + c];
    __shared__ float rs[256];
    rs[threadIdx.x] = s;
    __syncthreads();
    for (int o = blockDim.x >> 1; o > 0; o >>= 1) {
        if ((int)threadIdx.x < o) rs[threadIdx.x] += rs[threadIdx.x + o];
        __syncthreads();
    }
    if (threadIdx.x == 0) mean[b * 64 + c] = rs[0] / (float)per;
}

// ---------------------------------------------------------------------------
// head: 64->256 relu -> 256->128, batch-BN over 8 samples, L2 normalize
// ---------------------------------------------------------------------------
__global__ void k_head(const float* __restrict__ hm, const float* __restrict__ w1,
                       const float* __restrict__ b1, const float* __restrict__ w2,
                       const float* __restrict__ b2, const float* __restrict__ bg,
                       const float* __restrict__ bb, float* __restrict__ out) {
    __shared__ float e1[8 * 256];
    __shared__ float e2[8 * 128];
    __shared__ float nrm[8];
    int tid = threadIdx.x;
    for (int j = tid; j < 256; j += blockDim.x) {
        const float* wr = w1 + j * 64;
        for (int i = 0; i < 8; ++i) {
            float s = b1[j];
            for (int k = 0; k < 64; ++k) s += hm[i * 64 + k] * wr[k];
            e1[i * 256 + j] = fmaxf(s, 0.f);
        }
    }
    __syncthreads();
    for (int j = tid; j < 128; j += blockDim.x) {
        const float* wr = w2 + j * 256;
        for (int i = 0; i < 8; ++i) {
            float s = b2[j];
            for (int k = 0; k < 256; ++k) s += e1[i * 256 + k] * wr[k];
            e2[i * 128 + j] = s;
        }
    }
    __syncthreads();
    for (int j = tid; j < 128; j += blockDim.x) {
        float s = 0.f, ss = 0.f;
        for (int i = 0; i < 8; ++i) { float v = e2[i * 128 + j]; s += v; ss += v * v; }
        float m = s * 0.125f, var = ss * 0.125f - m * m;
        float inv = rsqrtf(var + EPSF);
        for (int i = 0; i < 8; ++i)
            e2[i * 128 + j] = (e2[i * 128 + j] - m) * inv * bg[j] + bb[j];
    }
    __syncthreads();
    if (tid < 8) {
        float s = 0.f;
        for (int k = 0; k < 128; ++k) { float v = e2[tid * 128 + k]; s += v * v; }
        nrm[tid] = fmaxf(sqrtf(s), 1e-12f);
    }
    __syncthreads();
    for (int j = tid; j < 128; j += blockDim.x)
        for (int i = 0; i < 8; ++i) out[i * 128 + j] = e2[i * 128 + j] / nrm[i];
}

// ---------------------------------------------------------------------------
// host-side orchestration
// ---------------------------------------------------------------------------
static void run_mamba(const float* xln, const float* inw, const float* cw,
                      const float* cb, const float* xw, const float* dtw,
                      const float* dtb, const float* Aneg, const float* Dp,
                      const float* outw, float* xcbuf, float* zbuf, float* projb,
                      float* hout, int nseq, int L, hipStream_t stream) {
    dim3 g128(NTOK / 128, D_INNER / 16);
    // xr = xln @ Win[0:128]^T ; z = xln @ Win[128:256]^T
    k_gemm_wmma<<<g128, 256, 0, stream>>>(xln, inw,                 xcbuf, NTOK, D_INNER, D_MODEL);
    k_gemm_wmma<<<g128, 256, 0, stream>>>(xln, inw + 128 * D_MODEL, zbuf,  NTOK, D_INNER, D_MODEL);
    k_dwconv<<<(nseq * D_INNER) / 256, 256, 0, stream>>>(xcbuf, cw, cb, nseq, L);
    k_xproj<<<NTOK / 8, 256, 0, stream>>>(xcbuf, xw, projb);
    k_scan<<<nseq, 128, 0, stream>>>(xcbuf, projb, zbuf, Aneg, Dp, dtw, dtb, L);
    k_gemm_wmma<<<dim3(NTOK / 128, D_MODEL / 16), 256, 0, stream>>>(xcbuf, outw, hout,
                                                                    NTOK, D_MODEL, D_INNER);
}

extern "C" void kernel_launch(void* const* d_in, const int* in_sizes, int n_in,
                              void* d_out, int out_size, void* d_ws, size_t ws_size,
                              hipStream_t stream) {
    (void)in_sizes; (void)n_in; (void)out_size; (void)ws_size;
    // inputs, setup_inputs() order (mamba dicts flattened in insertion order)
    const float* x      = (const float*)d_in[0];
    const float* conv1w = (const float*)d_in[1];
    const float* bn1g   = (const float*)d_in[2];
    const float* bn1b   = (const float*)d_in[3];
    const float* conv2w = (const float*)d_in[4];
    const float* bn2g   = (const float*)d_in[5];
    const float* bn2b   = (const float*)d_in[6];
    const float* lnfg   = (const float*)d_in[7];
    const float* lnfb   = (const float*)d_in[8];
    const float* f_inw  = (const float*)d_in[9];
    const float* f_cw   = (const float*)d_in[10];
    const float* f_cb   = (const float*)d_in[11];
    const float* f_xw   = (const float*)d_in[12];
    const float* f_dtw  = (const float*)d_in[13];
    const float* f_dtb  = (const float*)d_in[14];
    const float* f_alog = (const float*)d_in[15];
    const float* f_D    = (const float*)d_in[16];
    const float* f_outw = (const float*)d_in[17];
    const float* lntg   = (const float*)d_in[18];
    const float* lntb   = (const float*)d_in[19];
    const float* t_inw  = (const float*)d_in[20];
    const float* t_cw   = (const float*)d_in[21];
    const float* t_cb   = (const float*)d_in[22];
    const float* t_xw   = (const float*)d_in[23];
    const float* t_dtw  = (const float*)d_in[24];
    const float* t_dtb  = (const float*)d_in[25];
    const float* t_alog = (const float*)d_in[26];
    const float* t_D    = (const float*)d_in[27];
    const float* t_outw = (const float*)d_in[28];
    const float* hw1    = (const float*)d_in[29];
    const float* hb1    = (const float*)d_in[30];
    const float* hw2    = (const float*)d_in[31];
    const float* hb2    = (const float*)d_in[32];
    const float* bnhg   = (const float*)d_in[33];
    const float* bnhb   = (const float*)d_in[34];

    // workspace layout (floats)
    float* ws    = (float*)d_ws;
    float* A_f   = ws;                     // 2048
    float* A_t   = A_f + 2048;             // 2048
    float* wt2   = A_t + 2048;             // 9*64*32 = 18432 (prepacked conv2 w)
    float* sum1  = wt2 + 18432;            // 32
    float* sq1   = sum1 + 32;              // 32
    float* sc1   = sq1 + 32;               // 32
    float* sh1   = sc1 + 32;               // 32
    float* sum2  = sh1 + 32;               // 64
    float* sq2   = sum2 + 64;              // 64
    float* sc2   = sq2 + 64;               // 64
    float* sh2   = sc2 + 64;               // 64
    float* hmean = sh2 + 64;               // 512
    float* c1tok = hmean + 512;            // NTOK*32 (pixel-major, channel-last)
    float* c2tok = c1tok + (size_t)NTOK * 32;           // NTOK*64 (token-major)
    float* xln   = c2tok + (size_t)NTOK * 64;           // NTOK*64
    float* xcbuf = xln   + (size_t)NTOK * 64;           // NTOK*128 (xr -> xc -> y*silu(z))
    float* zbuf  = xcbuf + (size_t)NTOK * 128;          // NTOK*128
    float* projb = zbuf  + (size_t)NTOK * 128;          // NTOK*36
    float* hbuf  = projb + (size_t)NTOK * 36;           // NTOK*64

    // stage 0: constants
    k_zero<<<4, 256, 0, stream>>>(sum1, 896);
    k_negexp<<<8, 256, 0, stream>>>(A_f, f_alog, 2048);
    k_negexp<<<8, 256, 0, stream>>>(A_t, t_alog, 2048);
    k_packw2<<<72, 256, 0, stream>>>(conv2w, wt2);

    // stage 1: conv1 (channel-last) + bn1 stats
    k_conv1<<<(NTOK * 32) / 256, 256, 0, stream>>>(x, conv1w, c1tok);
    k_stats_tok<<<32, 256, 0, stream>>>(c1tok, 32, NTOK, sum1, sq1);
    k_bnfin<<<1, 64, 0, stream>>>(sum1, sq1, 1.f / (float)NTOK, bn1g, bn1b, sc1, sh1, 32);

    // stage 2: conv2 (WMMA implicit GEMM, bn1+relu folded into operand gather)
    k_conv2_wmma<<<dim3(NTOK / 128, 4), 256, 0, stream>>>(c1tok, wt2, sc1, sh1, c2tok);
    k_stats_tok<<<64, 256, 0, stream>>>(c2tok, 64, NTOK, sum2, sq2);
    k_bnfin<<<1, 64, 0, stream>>>(sum2, sq2, 1.f / (float)NTOK, bn2g, bn2b, sc2, sh2, 64);
    k_bn2ln<<<NTOK / 8, 256, 0, stream>>>(c2tok, sc2, sh2, lnfg, lnfb, xln);

    // stage 3: freq-axis mamba (2048 sequences of L=128, contiguous tokens)
    run_mamba(xln, f_inw, f_cw, f_cb, f_xw, f_dtw, f_dtb, A_f, f_D, f_outw,
              xcbuf, zbuf, projb, hbuf, 2048, 128, stream);

    // stage 4: temporal-axis mamba (1024 sequences of L=256)
    k_ln<<<NTOK / 8, 256, 0, stream>>>(hbuf, lntg, lntb, xln);
    run_mamba(xln, t_inw, t_cw, t_cb, t_xw, t_dtw, t_dtb, A_t, t_D, t_outw,
              xcbuf, zbuf, projb, hbuf, 1024, 256, stream);

    // stage 5: global mean + head
    k_mean<<<dim3(8, 64), 256, 0, stream>>>(hbuf, hmean);
    k_head<<<1, 256, 0, stream>>>(hmean, hw1, hb1, hw2, hb2, bnhg, bnhb, (float*)d_out);
}
